// ConvOffset3d_15092515078410
// MI455X (gfx1250) — matrix operand order repaired and checked by reading, not compiled, not run on hardware
//
#include <hip/hip_runtime.h>

typedef __attribute__((ext_vector_type(16))) _Float16 v16h;
typedef __attribute__((ext_vector_type(8)))  float    v8f;
typedef __attribute__((ext_vector_type(2), aligned(4))) float f2a;

#define C_IN 64
#define GROUPS 8
#define CPG_ 8
#define DD 8
#define HH 56
#define WW 56
#define KV 27
#define OC 64
#define HWs (HH * WW)              // 3136
#define NTOT (DD * HH * WW)        // 25088
#define CH_STRIDE NTOT             // x channel stride (elements)
#define NT 64                      // output positions per block
#define WSLICE (OC * C_IN)         // 4096 weights per tap

// One-time pre-pass: w[o][c][k] f32 -> wt[k][o*64+c] f16 (221KB, L2-resident).
__global__ void transpose_weights_f16(const float* __restrict__ w,
                                      _Float16* __restrict__ wt) {
  const int idx = blockIdx.x * 256 + threadIdx.x;  // 27*4096 total
  if (idx >= KV * WSLICE) return;
  const int k = idx >> 12;      // / 4096
  const int e = idx & 4095;     // o*64 + c
  wt[idx] = (_Float16)w[e * KV + k];
}

// Fused deformable-conv3d. Per block: 64 out-ch x 64 positions.
// Double-buffered LDS; paired-x float2 gathers; f16 WMMA accumulation.
template <bool USE_WT>
__launch_bounds__(256)
__global__ void deform_conv3d_wmma(const float* __restrict__ x,
                                   const float* __restrict__ offset,
                                   const float* __restrict__ weight,
                                   const _Float16* __restrict__ wt,
                                   float* __restrict__ out) {
  __shared__ _Float16 lds_v[2][NT * C_IN];   // [buf][n][c]  2 x 8KB
  __shared__ _Float16 lds_w[2][OC * C_IN];   // [buf][o][c]  2 x 8KB

  const int tid  = threadIdx.x;
  const int lane = tid & 31;
  const int wave = tid >> 5;
  const int lm   = lane & 15;
  const int hl   = lane >> 4;
  const int n0   = blockIdx.x * NT;
  const int wm   = wave >> 2;    // m-tile pair {2wm, 2wm+1}
  const int nt   = wave & 3;     // n-tile 0..3

  v8f acc0 = {};
  v8f acc1 = {};

  // ---- tap-invariant per-point state (2 sample points per thread) ----
  float zb[2], yb[2], xbf[2];      // (coord - 1) base per axis, as float
  const float* opb[2];             // offset ptr for tap 0
  const float* xgp[2];             // x base for this point's group
  int vsoff[2];                    // LDS store offset (halfs)
#pragma unroll
  for (int j = 0; j < 2; ++j) {
    const int p  = tid + 256 * j;
    const int g  = p >> 6;               // group 0..7
    const int nl = p & 63;               // local position
    const int n  = n0 + nl;
    const int dz = n / HWs;
    const int rm = n - dz * HWs;
    const int hy = rm / WW;
    const int wx = rm - hy * WW;
    zb[j]  = (float)(dz - 1);
    yb[j]  = (float)(hy - 1);
    xbf[j] = (float)(wx - 1);
    opb[j] = offset + (long)(g * KV * 3) * NTOT + n;
    xgp[j] = x + (long)(g * CPG_) * CH_STRIDE;
    vsoff[j] = nl * C_IN + g * CPG_;
  }

  // ---- stage weight slice for tap k into wbuf ----
  auto stage_w = [&](int k, _Float16* wbuf) {
    if (USE_WT) {
      const uint4* src = (const uint4*)(wt + (size_t)k * WSLICE) + tid * 2;
      uint4* dst = ((uint4*)wbuf) + tid * 2;
      dst[0] = src[0];
      dst[1] = src[1];
    } else {
#pragma unroll
      for (int i = 0; i < 16; ++i) {
        const int e = tid * 16 + i;
        wbuf[e] = (_Float16)weight[e * KV + k];
      }
    }
  };

  // ---- sample tap k: 512 points (g,n) over 256 threads -> vbuf[n][c] ----
  auto sample_tap = [&](int k, _Float16* vbuf) {
    const int kz = k / 9, ky = (k / 3) % 3, kx = k % 3;
    const float kzf = (float)kz, kyf = (float)ky, kxf = (float)kx;
#pragma unroll
    for (int j = 0; j < 2; ++j) {
      const float* op = opb[j] + (long)k * (3 * NTOT);
      const float oz = __builtin_nontemporal_load(op);
      const float oy = __builtin_nontemporal_load(op + NTOT);
      const float ox = __builtin_nontemporal_load(op + 2 * NTOT);

      const float zc = kzf + zb[j] + oz;
      const float yc = kyf + yb[j] + oy;
      const float xc = kxf + xbf[j] + ox;
      const float z0f = floorf(zc), y0f = floorf(yc), x0f = floorf(xc);
      const float fz = zc - z0f, fy = yc - y0f, fx = xc - x0f;
      const int z0 = (int)z0f, y0 = (int)y0f, x0 = (int)x0f;

      const float wz[2] = {1.f - fz, fz};
      const float wy[2] = {1.f - fy, fy};
      const float wx0 = 1.f - fx, wx1 = fx;
      const int zi[2] = {z0, z0 + 1};
      const int yi[2] = {y0, y0 + 1};
      const bool zv[2] = {(z0 >= 0) && (z0 < DD), (z0 >= -1) && (z0 < DD - 1)};
      const bool yv[2] = {(y0 >= 0) && (y0 < HH), (y0 >= -1) && (y0 < HH - 1)};
      const bool xv0 = (x0 >= 0) && (x0 < WW);
      const bool xv1 = (x0 >= -1) && (x0 < WW - 1);
      const int  xb = min(max(x0, 0), WW - 2);
      const bool swapx = (x0 != xb);   // pair base clamped: swap lanes

      float cw0[4], cw1[4];
      int   cb4[4];
#pragma unroll
      for (int a = 0; a < 2; ++a)
#pragma unroll
        for (int b = 0; b < 2; ++b) {
          const int q = a * 2 + b;
          const float wzy = wz[a] * wy[b];
          const bool v = zv[a] && yv[b];
          cw0[q] = (v && xv0) ? wzy * wx0 : 0.f;
          cw1[q] = (v && xv1) ? wzy * wx1 : 0.f;
          const int zcl = min(max(zi[a], 0), DD - 1);
          const int ycl = min(max(yi[b], 0), HH - 1);
          cb4[q] = zcl * HWs + ycl * WW + xb;
        }

#pragma unroll
      for (int ch = 0; ch < CPG_; ++ch) {
        const float* xc_ = xgp[j] + (long)ch * CH_STRIDE;
        float s = 0.f;
#pragma unroll
        for (int q = 0; q < 4; ++q) {
          const f2a v2 = *(const f2a*)(xc_ + cb4[q]);
          const float e0 = swapx ? v2.y : v2.x;
          const float e1 = swapx ? v2.x : v2.y;
          s = fmaf(cw0[q], e0, s);
          s = fmaf(cw1[q], e1, s);
        }
        vbuf[vsoff[j] + ch] = (_Float16)s;
      }
    }
  };

  // prologue: stage tap 0
  stage_w(0, lds_w[0]);
  sample_tap(0, lds_v[0]);
  __syncthreads();

  for (int k = 0; k < KV; ++k) {
    const int cur = k & 1;
    const int nxt = cur ^ 1;

    // stage next tap into the other buffers (overlaps with WMMA below)
    if (k + 1 < KV) {
      stage_w(k + 1, lds_w[nxt]);
      sample_tap(k + 1, lds_v[nxt]);
    }

    // ---- WMMA: out_tile += W_k (A, M=o K=c) x val_k (B, K=c N=n) ----
#pragma unroll
    for (int kh = 0; kh < 2; ++kh) {
      const int cbase = kh * 32;

      // B fragment: b[i] = val[K = hl*16 + i][N = nt*16 + lm]
      v16h bf;
      const _Float16* bp = &lds_v[cur][(nt * 16 + lm) * C_IN + cbase + hl * 16];
#pragma unroll
      for (int i = 0; i < 16; ++i) bf[i] = bp[i];

      // A fragments: a[i] = W[M = o][K = hl*16 + i]
#pragma unroll
      for (int t = 0; t < 2; ++t) {
        const int o = (wm * 2 + t) * 16 + lm;
        const _Float16* ap = &lds_w[cur][o * C_IN + cbase];
        v16h af;
#pragma unroll
        for (int i = 0; i < 16; ++i) af[i] = ap[hl * 16 + i];
        if (t == 0)
          acc0 = __builtin_amdgcn_wmma_f32_16x16x32_f16(
              false, af, false, bf, (short)0, acc0, false, false);
        else
          acc1 = __builtin_amdgcn_wmma_f32_16x16x32_f16(
              false, af, false, bf, (short)0, acc1, false, false);
      }
    }
    __syncthreads();
  }

  // ---- writeback (non-temporal): C/D layout M = hl*8 + r, N = lm ----
  const int n = n0 + nt * 16 + lm;
#pragma unroll
  for (int r = 0; r < 8; ++r) {
    const int o0 = (wm * 2 + 0) * 16 + hl * 8 + r;
    const int o1 = (wm * 2 + 1) * 16 + hl * 8 + r;
    __builtin_nontemporal_store(acc0[r], &out[(long)o0 * NTOT + n]);
    __builtin_nontemporal_store(acc1[r], &out[(long)o1 * NTOT + n]);
  }
}

extern "C" void kernel_launch(void* const* d_in, const int* in_sizes, int n_in,
                              void* d_out, int out_size, void* d_ws, size_t ws_size,
                              hipStream_t stream) {
  (void)in_sizes; (void)n_in; (void)out_size;
  const float* x      = (const float*)d_in[0];
  const float* offset = (const float*)d_in[1];
  const float* weight = (const float*)d_in[2];
  float* out = (float*)d_out;

  const size_t wt_bytes = (size_t)KV * WSLICE * sizeof(_Float16);  // 221184
  const dim3 grid(NTOT / NT);  // 392 blocks

  if (ws_size >= wt_bytes) {
    _Float16* wtp = (_Float16*)d_ws;
    hipLaunchKernelGGL(transpose_weights_f16, dim3((KV * WSLICE + 255) / 256),
                       dim3(256), 0, stream, weight, wtp);
    hipLaunchKernelGGL(deform_conv3d_wmma<true>, grid, dim3(256), 0, stream,
                       x, offset, weight, wtp, out);
  } else {
    hipLaunchKernelGGL(deform_conv3d_wmma<false>, grid, dim3(256), 0, stream,
                       x, offset, weight, nullptr, out);
  }
}
